// MultiHeadAttention_67250597920960
// MI455X (gfx1250) — matrix operand order repaired and checked by reading, not compiled, not run on hardware
//
#include <hip/hip_runtime.h>
#include <hip/hip_bf16.h>
#include <math.h>
#include <stdint.h>

typedef __bf16 bf16_t;
typedef bf16_t v16bf __attribute__((ext_vector_type(16)));
typedef bf16_t v8bf  __attribute__((ext_vector_type(8)));
typedef float  v8f   __attribute__((ext_vector_type(8)));
typedef int    v4i   __attribute__((ext_vector_type(4)));

#define WMMA_BF16(a, b, c) \
  __builtin_amdgcn_wmma_f32_16x16x32_bf16(false, (a), false, (b), (short)0, (c), false, false)

#ifndef __has_builtin
#define __has_builtin(x) 0
#endif
#if __has_builtin(__builtin_amdgcn_global_load_async_to_lds_b128)
#define HAVE_ASYNC 1
#else
#define HAVE_ASYNC 0
#endif
#if __has_builtin(__builtin_amdgcn_s_wait_asynccnt)
#define HAVE_WAIT_ASYNC 1
#else
#define HAVE_WAIT_ASYNC 0
#endif

static constexpr int Bc  = 2;
static constexpr int Tc  = 2048;
static constexpr int Cc  = 2048;
static constexpr int Hc  = 16;
static constexpr int Dc  = 128;
static constexpr int NTOK = Bc * Tc;        // 4096 tokens
static constexpr int NCAT = 5120;           // q 2048 | k 512 | v 512 | gate 2048
static constexpr int COL_K = 2048, COL_V = 2560, COL_G = 3072;

// ---- workspace layout (bytes) ----
static constexpr size_t OFF_XB   = 0;                                   // bf16 [4096,2048]
static constexpr size_t OFF_WCT  = OFF_XB  + (size_t)NTOK * Cc * 2;     // bf16 [5120,2048] (W^T cat)
static constexpr size_t OFF_QKVG = OFF_WCT + (size_t)NCAT * Cc * 2;     // f32  [4096,5120]
static constexpr size_t OFF_QB   = OFF_QKVG+ (size_t)NTOK * NCAT * 4;   // bf16 [4096,2048]
static constexpr size_t OFF_KB   = OFF_QB  + (size_t)NTOK * 2048 * 2;   // bf16 [4096,512]
static constexpr size_t OFF_VT   = OFF_KB  + (size_t)NTOK * 512 * 2;    // bf16 [2*512, 2048] (V^T)
static constexpr size_t OFF_YB   = OFF_VT  + (size_t)Bc * 512 * Tc * 2; // bf16 [4096,2048]
static constexpr size_t OFF_WPT  = OFF_YB  + (size_t)NTOK * 2048 * 2;   // bf16 [2048,2048] (w_proj^T)

__device__ __forceinline__ bf16_t f2b(float f) {
  union { float f; unsigned int u; } v; v.f = f;
  unsigned int u = v.u + 0x7FFFu + ((v.u >> 16) & 1u);   // round-to-nearest-even
  union { unsigned short s; bf16_t b; } o; o.s = (unsigned short)(u >> 16);
  return o.b;
}

// 16-byte global -> LDS copy: async on CDNA5, sync fallback otherwise
__device__ __forceinline__ void cp16_g2l(const bf16_t* g, bf16_t* l) {
#if HAVE_ASYNC
  __builtin_amdgcn_global_load_async_to_lds_b128((v4i*)g, (v4i*)l, 0, 0);
#else
  *(v8bf*)l = *(const v8bf*)g;
#endif
}

template <int N>
__device__ __forceinline__ void wait_async() {
#if HAVE_ASYNC
#if HAVE_WAIT_ASYNC
  __builtin_amdgcn_s_wait_asynccnt(N);
#else
  if constexpr (N == 0) asm volatile("s_wait_asynccnt 0x0" ::: "memory");
  else                  asm volatile("s_wait_asynccnt 0x4" ::: "memory");
#endif
#endif
}

// -------------------- converters --------------------
__global__ void cvt_f32_bf16(const float* __restrict__ src, bf16_t* __restrict__ dst, int n) {
  int i = blockIdx.x * blockDim.x + threadIdx.x;
  if (i < n) dst[i] = f2b(src[i]);
}

// src[K,N] f32 row-major -> dst[N,K] bf16 row-major (transpose+convert)
__global__ void cvt_transpose(const float* __restrict__ src, bf16_t* __restrict__ dst, int K, int N) {
  int i = blockIdx.x * blockDim.x + threadIdx.x;
  if (i < K * N) {
    int k = i / N, n = i - k * N;
    dst[(size_t)n * K + k] = f2b(src[i]);
  }
}

// -------------------- WMMA GEMM: C[M,N] = A[M,K] * Bt[N,K]^T --------------------
// 128 threads = 4 waves; block tile 64(M) x 64(N); wave w owns rows [w*16, w*16+16).
// A/B k-slices (64x32 bf16 each) staged in LDS, double-buffered via async copies.
__global__ __launch_bounds__(128) void gemm_bf16_wmma_lds(
    const bf16_t* __restrict__ A, const bf16_t* __restrict__ Bt,
    float* __restrict__ C, int K, int ldc) {
  const int n0   = blockIdx.x * 64;
  const int m0   = blockIdx.y * 64;
  const int tid  = threadIdx.x;
  const int wv   = tid >> 5;
  const int lane = tid & 31;
  const int half = lane >> 4;
  const int l16  = lane & 15;

  __shared__ __align__(32) bf16_t Ab[2][64 * 32];
  __shared__ __align__(32) bf16_t Bb[2][64 * 32];

  // each thread copies two 16B chunks of A and two of B per k-step
  const int c0 = tid * 2, c1 = tid * 2 + 1;   // chunk ids, row = c>>2, koff = (c&3)*8

  const bf16_t* Ag0 = A  + (size_t)(m0 + (c0 >> 2)) * K + (c0 & 3) * 8;
  const bf16_t* Ag1 = A  + (size_t)(m0 + (c1 >> 2)) * K + (c1 & 3) * 8;
  const bf16_t* Bg0 = Bt + (size_t)(n0 + (c0 >> 2)) * K + (c0 & 3) * 8;
  const bf16_t* Bg1 = Bt + (size_t)(n0 + (c1 >> 2)) * K + (c1 & 3) * 8;

  v8f acc0 = {}, acc1 = {}, acc2 = {}, acc3 = {};
  const int KT = K >> 5;

  // prologue: fill buffer 0
  cp16_g2l(Ag0, &Ab[0][c0 * 8]);
  cp16_g2l(Ag1, &Ab[0][c1 * 8]);
  cp16_g2l(Bg0, &Bb[0][c0 * 8]);
  cp16_g2l(Bg1, &Bb[0][c1 * 8]);

  for (int kt = 0; kt < KT; ++kt) {
    if (kt + 1 < KT) {
      const int nb = (kt + 1) & 1, k1 = (kt + 1) << 5;
      cp16_g2l(Ag0 + k1, &Ab[nb][c0 * 8]);
      cp16_g2l(Ag1 + k1, &Ab[nb][c1 * 8]);
      cp16_g2l(Bg0 + k1, &Bb[nb][c0 * 8]);
      cp16_g2l(Bg1 + k1, &Bb[nb][c1 * 8]);
      wait_async<4>();          // current buffer's 4 copies complete (in-order)
    } else {
      wait_async<0>();
    }
    __syncthreads();

    const bf16_t* ab = Ab[kt & 1];
    const bf16_t* bb = Bb[kt & 1];
    const bf16_t* ar = ab + (wv * 16 + l16) * 32 + half * 16;
    v8bf a0 = *(const v8bf*)(ar);
    v8bf a1 = *(const v8bf*)(ar + 8);
    v16bf af = __builtin_shufflevector(a0, a1, 0,1,2,3,4,5,6,7,8,9,10,11,12,13,14,15);
    v16bf b0 = *(const v16bf*)(bb + ( 0 + l16) * 32 + half * 16);
    v16bf b1 = *(const v16bf*)(bb + (16 + l16) * 32 + half * 16);
    v16bf b2 = *(const v16bf*)(bb + (32 + l16) * 32 + half * 16);
    v16bf b3 = *(const v16bf*)(bb + (48 + l16) * 32 + half * 16);
    acc0 = WMMA_BF16(af, b0, acc0);
    acc1 = WMMA_BF16(af, b1, acc1);
    acc2 = WMMA_BF16(af, b2, acc2);
    acc3 = WMMA_BF16(af, b3, acc3);
    __syncthreads();
  }

#pragma unroll
  for (int r = 0; r < 8; ++r) {
    size_t m = (size_t)(m0 + wv * 16 + r + half * 8);
    C[m * ldc + n0 +  0 + l16] = acc0[r];
    C[m * ldc + n0 + 16 + l16] = acc1[r];
    C[m * ldc + n0 + 32 + l16] = acc2[r];
    C[m * ldc + n0 + 48 + l16] = acc3[r];
  }
}

// -------------------- RMS-norm + RoPE + convert epilogue --------------------
__global__ __launch_bounds__(128) void norm_rope_cvt(
    const float* __restrict__ QKVG, const float* __restrict__ cosp, const float* __restrict__ sinp,
    const float* __restrict__ qw, const float* __restrict__ kw,
    bf16_t* __restrict__ Qb, bf16_t* __restrict__ Kb, bf16_t* __restrict__ Vt) {
  const int which = blockIdx.x;
  const int token = blockIdx.y;
  const int d = threadIdx.x;
  const int b = token >> 11, t = token & (Tc - 1);
  const float* row = QKVG + (size_t)token * NCAT;
  __shared__ float red[128];
  __shared__ float xn[128];

  if (which < 20) {
    const bool isq = which < 16;
    const int h = isq ? which : which - 16;
    float x = isq ? row[h * Dc + d] : row[COL_K + h * Dc + d];
    red[d] = x * x;
    __syncthreads();
    for (int s = 64; s > 0; s >>= 1) { if (d < s) red[d] += red[d + s]; __syncthreads(); }
    float inv = rsqrtf(red[0] * (1.0f / 128.0f) + 1e-6f);
    xn[d] = x * inv * (isq ? qw[d] : kw[d]);
    __syncthreads();
    float y;
    if (d < 64) {
      float c = cosp[t * 64 + d], s = sinp[t * 64 + d];
      y = xn[d] * c - xn[d + 64] * s;
    } else {
      float c = cosp[t * 64 + d - 64], s = sinp[t * 64 + d - 64];
      y = xn[d - 64] * s + xn[d] * c;
    }
    if (isq) {
      y *= 0.08838834764831845f;  // fold 1/sqrt(128) into Q
      Qb[(size_t)token * 2048 + h * Dc + d] = f2b(y);
    } else {
      Kb[(size_t)token * 512 + h * Dc + d] = f2b(y);
    }
  } else {
    const int kh = which - 20;
    float x = row[COL_V + kh * Dc + d];
    Vt[(size_t)(b * 512 + kh * Dc + d) * Tc + t] = f2b(x);  // V transposed: [d][s]
  }
}

// -------------------- flash attention (one wave = one (b,h,16-row) tile) --------------------
__global__ __launch_bounds__(32) void attn_wmma(
    const bf16_t* __restrict__ Qb, const bf16_t* __restrict__ Kb, const bf16_t* __restrict__ Vt,
    const float* __restrict__ QKVG, bf16_t* __restrict__ Yb) {
  const int bid = blockIdx.x;
  const int mt = bid & 127;
  const int h  = (bid >> 7) & 15;
  const int b  = bid >> 11;
  const int kvh = h >> 2;
  const int lane = threadIdx.x, half = lane >> 4, l16 = lane & 15;
  const int t0 = mt * 16;

  v16bf qf[4];
  const bf16_t* qrow = Qb + (size_t)(b * Tc + t0 + l16) * 2048 + h * Dc + half * 16;
#pragma unroll
  for (int kc = 0; kc < 4; ++kc) {
    v8bf q0 = *(const v8bf*)(qrow + kc * 32);
    v8bf q1 = *(const v8bf*)(qrow + kc * 32 + 8);
    qf[kc] = __builtin_shufflevector(q0, q1, 0,1,2,3,4,5,6,7,8,9,10,11,12,13,14,15);
  }

  float rowmax[8], rowsum[8];
  v8f o[8];
#pragma unroll
  for (int r = 0; r < 8; ++r) { rowmax[r] = -1e30f; rowsum[r] = 0.0f; o[r] = (v8f){}; }

  __shared__ __align__(32) bf16_t pls[16 * 32];

  for (int s0 = 0; s0 < Tc; s0 += 32) {
    v8f s[2]; s[0] = (v8f){}; s[1] = (v8f){};
#pragma unroll
    for (int j = 0; j < 2; ++j) {
      const bf16_t* krow = Kb + (size_t)(b * Tc + s0 + j * 16 + l16) * 512 + kvh * Dc + half * 16;
#pragma unroll
      for (int kc = 0; kc < 4; ++kc) {
        v16bf kf = *(const v16bf*)(krow + kc * 32);
        s[j] = WMMA_BF16(qf[kc], kf, s[j]);
      }
    }
    float pj0[8], pj1[8];
#pragma unroll
    for (int r = 0; r < 8; ++r) {
      float mx = fmaxf(s[0][r], s[1][r]);
      for (int off = 8; off >= 1; off >>= 1) mx = fmaxf(mx, __shfl_xor(mx, off, 32));
      float mnew = fmaxf(rowmax[r], mx);
      float corr = __expf(rowmax[r] - mnew);
      rowmax[r] = mnew;
      float p0 = __expf(s[0][r] - mnew);
      float p1 = __expf(s[1][r] - mnew);
      float ps = p0 + p1;
      for (int off = 8; off >= 1; off >>= 1) ps += __shfl_xor(ps, off, 32);
      rowsum[r] = rowsum[r] * corr + ps;
      pj0[r] = p0; pj1[r] = p1;
#pragma unroll
      for (int ot = 0; ot < 8; ++ot) o[ot][r] *= corr;
    }
#pragma unroll
    for (int r = 0; r < 8; ++r) {
      int m = r + half * 8;
      pls[m * 32 + l16]      = f2b(pj0[r]);
      pls[m * 32 + 16 + l16] = f2b(pj1[r]);
    }
    __syncthreads();
    v8bf p0 = *(const v8bf*)(pls + l16 * 32 + half * 16);
    v8bf p1 = *(const v8bf*)(pls + l16 * 32 + half * 16 + 8);
    v16bf pf = __builtin_shufflevector(p0, p1, 0,1,2,3,4,5,6,7,8,9,10,11,12,13,14,15);
    __syncthreads();
#pragma unroll
    for (int ot = 0; ot < 8; ++ot) {
      const bf16_t* vrow = Vt + (size_t)(b * 512 + kvh * Dc + ot * 16 + l16) * Tc + s0 + half * 16;
      v16bf vf = *(const v16bf*)vrow;
      o[ot] = WMMA_BF16(pf, vf, o[ot]);
    }
  }

  const float* grow = QKVG + COL_G + (size_t)h * Dc;
#pragma unroll
  for (int r = 0; r < 8; ++r) {
    int m = r + half * 8;
    size_t tok = (size_t)(b * Tc + t0 + m);
    float inv = 1.0f / rowsum[r];
#pragma unroll
    for (int ot = 0; ot < 8; ++ot) {
      int d = ot * 16 + l16;
      float g = grow[tok * NCAT + d];
      float y = o[ot][r] * inv * (1.0f / (1.0f + __expf(-g)));
      Yb[tok * 2048 + h * Dc + d] = f2b(y);
    }
  }
}

extern "C" void kernel_launch(void* const* d_in, const int* in_sizes, int n_in,
                              void* d_out, int out_size, void* d_ws, size_t ws_size,
                              hipStream_t stream) {
  (void)in_sizes; (void)n_in; (void)out_size; (void)ws_size;
  const float* x     = (const float*)d_in[0];
  const float* cosp  = (const float*)d_in[1];
  const float* sinp  = (const float*)d_in[2];
  const float* wq    = (const float*)d_in[3];
  const float* wk    = (const float*)d_in[4];
  const float* wv    = (const float*)d_in[5];
  const float* wg    = (const float*)d_in[6];
  const float* wproj = (const float*)d_in[7];
  const float* qw    = (const float*)d_in[8];
  const float* kw    = (const float*)d_in[9];

  char* ws = (char*)d_ws;
  bf16_t* Xb   = (bf16_t*)(ws + OFF_XB);
  bf16_t* WCT  = (bf16_t*)(ws + OFF_WCT);
  float*  QKVG = (float*) (ws + OFF_QKVG);
  bf16_t* Qb   = (bf16_t*)(ws + OFF_QB);
  bf16_t* Kb   = (bf16_t*)(ws + OFF_KB);
  bf16_t* Vt   = (bf16_t*)(ws + OFF_VT);
  bf16_t* Yb   = (bf16_t*)(ws + OFF_YB);
  bf16_t* WPT  = (bf16_t*)(ws + OFF_WPT);

  {
    int n = NTOK * Cc;
    cvt_f32_bf16<<<(n + 255) / 256, 256, 0, stream>>>(x, Xb, n);
  }
  cvt_transpose<<<(Cc * 2048 + 255) / 256, 256, 0, stream>>>(wq,    WCT,                      Cc, 2048);
  cvt_transpose<<<(Cc * 512  + 255) / 256, 256, 0, stream>>>(wk,    WCT + (size_t)COL_K * Cc, Cc, 512);
  cvt_transpose<<<(Cc * 512  + 255) / 256, 256, 0, stream>>>(wv,    WCT + (size_t)COL_V * Cc, Cc, 512);
  cvt_transpose<<<(Cc * 2048 + 255) / 256, 256, 0, stream>>>(wg,    WCT + (size_t)COL_G * Cc, Cc, 2048);
  cvt_transpose<<<(2048 * 2048 + 255) / 256, 256, 0, stream>>>(wproj, WPT,                    2048, 2048);

  // fused QKV+gate projection: [4096,2048] x [2048,5120] -> f32 [4096,5120]
  gemm_bf16_wmma_lds<<<dim3(NCAT / 64, NTOK / 64), 128, 0, stream>>>(Xb, WCT, QKVG, Cc, NCAT);

  norm_rope_cvt<<<dim3(24, NTOK), 128, 0, stream>>>(QKVG, cosp, sinp, qw, kw, Qb, Kb, Vt);

  attn_wmma<<<Bc * Hc * (Tc / 16), 32, 0, stream>>>(Qb, Kb, Vt, QKVG, Yb);

  // output projection: [4096,2048] x [2048,2048] -> d_out f32
  gemm_bf16_wmma_lds<<<dim3(Cc / 64, NTOK / 64), 128, 0, stream>>>(Yb, WPT, (float*)d_out, 2048, Cc);
}
